// LocalMPAttentionBlock_51651276701870
// MI455X (gfx1250) — compile-verified, matrix operand order checked by reading, not run on hardware
//
#include <hip/hip_runtime.h>
#include <hip/hip_bf16.h>
#include <math.h>

// ---------------------------------------------------------------------------
// LocalMP + edge-softmax attention block for MI455X (gfx1250, wave32, WMMA)
// D = 128 fixed. GEMMs use v_wmma_f32_16x16x32_bf16 (bf16 in, f32 acc).
// ---------------------------------------------------------------------------

#define DIM 128

typedef __bf16 v16bf __attribute__((ext_vector_type(16)));
typedef float  v8f   __attribute__((ext_vector_type(8)));
typedef unsigned int v8u __attribute__((ext_vector_type(8)));

union FragBF { v8u u; v16bf b; };

__device__ __forceinline__ unsigned short f32_to_bf16_bits(float f) {
    // round-to-nearest-even bf16
    unsigned int x = __float_as_uint(f);
    unsigned int rounding = 0x7FFFu + ((x >> 16) & 1u);
    return (unsigned short)((x + rounding) >> 16);
}

__device__ __forceinline__ unsigned int pack_bf16x2(float lo, float hi) {
    return (unsigned int)f32_to_bf16_bits(lo) |
           ((unsigned int)f32_to_bf16_bits(hi) << 16);
}

__device__ __forceinline__ void atomicMaxFloat(float* addr, float val) {
    // works with -inf init: sign-aware monotone bit trick
    if (val >= 0.0f) atomicMax((int*)addr, __float_as_int(val));
    else             atomicMin((unsigned int*)addr, __float_as_uint(val));
}

// ---------------------------------------------------------------------------
// One-shot weight prep: 6 matrices -> bf16, transposed to [col j][k] layout.
// ---------------------------------------------------------------------------
__global__ void weights_to_bf16t(const float* __restrict__ W0,
                                 const float* __restrict__ W1,
                                 const float* __restrict__ W2,
                                 const float* __restrict__ W3,
                                 const float* __restrict__ W4,
                                 const float* __restrict__ W5,
                                 unsigned short* __restrict__ out)
{
    int i = blockIdx.x * blockDim.x + threadIdx.x;
    if (i >= 6 * DIM * DIM) return;
    int m = i >> 14;          // which matrix
    int r = i & (DIM * DIM - 1);
    int j = r >> 7, k = r & 127;
    const float* W;
    switch (m) {
        case 0: W = W0; break; case 1: W = W1; break;
        case 2: W = W2; break; case 3: W = W3; break;
        case 4: W = W4; break; default: W = W5; break;
    }
    out[i] = f32_to_bf16_bits(W[(size_t)k * DIM + j]);   // out[j][k] = W[k][j]
}

// ---------------------------------------------------------------------------
// Tile compute body, templated on uniform predicates so the hot loop carries
// zero per-element branching. One wave computes one 16-row tile across the
// 8 column tiles; 4x v_wmma_f32_16x16x32_bf16 per tile (K = 128).
// ---------------------------------------------------------------------------
template <bool FULL, bool HAS_R>
__device__ __forceinline__ void gemm_compute_128(
    const unsigned short* __restrict__ As,   // LDS [row][k] bf16
    const unsigned short* __restrict__ Bs,   // LDS [col j][k] bf16
    const float* __restrict__ Rp,
    float* __restrict__ Y,
    int row0, int n, int wave, int lane)
{
    const int m16  = lane & 15;    // M (A frag) / N (B,C frags)
    const int half = lane >> 4;
    const int arow = wave * 16 + m16;

    for (int ct = 0; ct < 8; ++ct) {
        v8f acc;
        #pragma unroll
        for (int i = 0; i < 8; ++i) {
            int gr = row0 + wave * 16 + i + 8 * half;     // C layout: M = i + 8*half
            if (HAS_R) {
                if (FULL)           acc[i] = Rp[(size_t)gr * DIM + ct * 16 + m16];
                else                acc[i] = (gr < n) ? Rp[(size_t)gr * DIM + ct * 16 + m16] : 0.0f;
            } else {
                acc[i] = 0.0f;
            }
        }
        #pragma unroll
        for (int kk = 0; kk < 4; ++kk) {
            const int k0 = kk * 32;
            FragBF a, b;
            #pragma unroll
            for (int j = 0; j < 8; ++j) {
                // 16-bit A 16x32 layout: j<4 -> K=2j+8*half ; j>=4 -> K=16+2(j-4)+8*half
                int kA = k0 + ((j < 4) ? (2 * j) : (16 + 2 * (j - 4))) + 8 * half;
                a.u[j] = *(const unsigned int*)&As[arow * 128 + kA];
                b.u[j] = *(const unsigned int*)&Bs[(ct * 16 + m16) * 128 + kA];
            }
            acc = __builtin_amdgcn_wmma_f32_16x16x32_bf16(
                      false, a.b, false, b.b, (short)0, acc, false, false);
        }
        #pragma unroll
        for (int i = 0; i < 8; ++i) {
            int gr = row0 + wave * 16 + i + 8 * half;
            if (FULL || gr < n) Y[(size_t)gr * DIM + ct * 16 + m16] = acc[i];
        }
    }
}

// ---------------------------------------------------------------------------
// Y_w[n,128] = X[n,128] @ W_w[128,128] (+ optional residual R on weight 0)
// Up to 3 weight/output pairs share one staged A block (QKV fusion).
// Weights arrive pre-transposed bf16 ([j][k]); staged into LDS as uint4 copy.
// ---------------------------------------------------------------------------
__launch_bounds__(256)
__global__ void gemm128_wmma_multi(const float* __restrict__ X,
                                   const unsigned short* __restrict__ Wt0,
                                   const unsigned short* __restrict__ Wt1,
                                   const unsigned short* __restrict__ Wt2,
                                   const float* __restrict__ R,   // residual for pair 0, may be null
                                   float* __restrict__ Y0,
                                   float* __restrict__ Y1,
                                   float* __restrict__ Y2,
                                   int n)
{
    __shared__ __align__(16) unsigned short As[128 * 128]; // [row][k] bf16
    __shared__ __align__(16) unsigned short Bs[128 * 128]; // [col j][k] bf16

    const int tid  = threadIdx.x;
    const int row0 = blockIdx.x * 128;
    const bool full = (row0 + 128 <= n);    // uniform: 156/157 blocks

    // ---- stage A block as bf16 ----
    if (full) {
        const float4* src = (const float4*)(X + (size_t)row0 * DIM);
        uint2* dst = (uint2*)As;
        for (int i = tid; i < 128 * 32; i += 256) {        // 4096 float4 -> uint2
            float4 f = src[i];
            dst[i] = make_uint2(pack_bf16x2(f.x, f.y), pack_bf16x2(f.z, f.w));
        }
    } else {
        for (int i = tid; i < 128 * 128; i += 256) {
            int r = i >> 7, c = i & 127;
            float fa = (row0 + r < n) ? X[(size_t)(row0 + r) * DIM + c] : 0.0f;
            As[i] = f32_to_bf16_bits(fa);
        }
    }

    const int wave = tid >> 5;
    const int lane = tid & 31;

    const unsigned short* Wts[3] = { Wt0, Wt1, Wt2 };
    float*                Ys[3]  = { Y0,  Y1,  Y2  };

    for (int wsel = 0; wsel < 3; ++wsel) {
        const unsigned short* Wt = Wts[wsel];
        if (Wt == nullptr) break;
        __syncthreads();   // A staged (iter 0) / prev compute done reading Bs
        {
            const uint4* w4 = (const uint4*)Wt;
            uint4* b4 = (uint4*)Bs;
            for (int i = tid; i < 2048; i += 256) b4[i] = w4[i];
        }
        __syncthreads();

        float* __restrict__ Y  = Ys[wsel];
        const float* __restrict__ Rp = (wsel == 0) ? R : nullptr;

        // uniform dispatch: hot loops carry no per-element predicates
        if (full) {
            if (Rp) gemm_compute_128<true,  true >(As, Bs, Rp,      Y, row0, n, wave, lane);
            else    gemm_compute_128<true,  false>(As, Bs, nullptr, Y, row0, n, wave, lane);
        } else {
            if (Rp) gemm_compute_128<false, true >(As, Bs, Rp,      Y, row0, n, wave, lane);
            else    gemm_compute_128<false, false>(As, Bs, nullptr, Y, row0, n, wave, lane);
        }
    }
}

// ---------------------------------------------------------------------------
// init: zero agg/out_agg/den, set mx = -inf
// ---------------------------------------------------------------------------
__global__ void init_ws_kernel(float* agg, float* oagg, float* den, float* mx,
                               int nND, int nN)
{
    int i = blockIdx.x * blockDim.x + threadIdx.x;
    if (i < nND) { agg[i] = 0.0f; oagg[i] = 0.0f; }
    if (i < nN)  { den[i] = 0.0f; mx[i] = -INFINITY; }
}

// ---------------------------------------------------------------------------
// LocalMP scatter: wave per edge, lane owns a float4 of the 128-wide row.
// agg[dst] += exp(-6*len) * hW[src]
// ---------------------------------------------------------------------------
__global__ void mp_scatter_kernel(const float* __restrict__ hW,
                                  const int* __restrict__ ei,
                                  const float* __restrict__ elen,
                                  float* __restrict__ agg, int E)
{
    long long t = (long long)blockIdx.x * blockDim.x + threadIdx.x;
    int e = (int)(t >> 5);
    if (e >= E) return;
    int lane = (int)(t & 31);
    int src = ei[e], dst = ei[E + e];
    float decay = __expf(-6.0f * elen[e]);
    float4 m = ((const float4*)(hW + (size_t)src * DIM))[lane];
    float* pd = agg + (size_t)dst * DIM + lane * 4;
    unsafeAtomicAdd(pd + 0, m.x * decay);
    unsafeAtomicAdd(pd + 1, m.y * decay);
    unsafeAtomicAdd(pd + 2, m.z * decay);
    unsafeAtomicAdd(pd + 3, m.w * decay);
}

// ---------------------------------------------------------------------------
// g[n,3] = q[n,:] . We[j,:]  (folds edge_vec @ We into a 3-FMA edge term)
// ---------------------------------------------------------------------------
__global__ void qWe_kernel(const float* __restrict__ q,
                           const float* __restrict__ We,
                           float* __restrict__ g, int n)
{
    int nn = blockIdx.x * blockDim.x + threadIdx.x;
    if (nn >= n) return;
    float s0 = 0.f, s1 = 0.f, s2 = 0.f;
    const float* qr = q + (size_t)nn * DIM;
    #pragma unroll 4
    for (int d = 0; d < DIM; ++d) {
        float qd = qr[d];
        s0 = fmaf(qd, We[d],           s0);
        s1 = fmaf(qd, We[DIM + d],     s1);
        s2 = fmaf(qd, We[2 * DIM + d], s2);
    }
    g[nn * 3 + 0] = s0; g[nn * 3 + 1] = s1; g[nn * 3 + 2] = s2;
}

// ---------------------------------------------------------------------------
// logits: wave per edge. logit = (q[dst].k[src] + g[dst].ev)/sqrt(D)
//                               - softplus(decay_scale)*len ; atomic max per dst
// ---------------------------------------------------------------------------
__global__ void edge_logits_kernel(const float* __restrict__ q,
                                   const float* __restrict__ kf,
                                   const float* __restrict__ g,
                                   const int* __restrict__ ei,
                                   const float* __restrict__ evec,
                                   const float* __restrict__ elen,
                                   const float* __restrict__ decay_scale,
                                   float* __restrict__ logits,
                                   float* __restrict__ mx, int E)
{
    long long t = (long long)blockIdx.x * blockDim.x + threadIdx.x;
    int e = (int)(t >> 5);
    if (e >= E) return;
    int lane = (int)(t & 31);
    int src = ei[e], dst = ei[E + e];
    float4 qa = ((const float4*)(q  + (size_t)dst * DIM))[lane];
    float4 ka = ((const float4*)(kf + (size_t)src * DIM))[lane];
    float p = qa.x * ka.x + qa.y * ka.y + qa.z * ka.z + qa.w * ka.w;
    #pragma unroll
    for (int o = 16; o > 0; o >>= 1) p += __shfl_xor(p, o, 32);
    if (lane == 0) {
        float et = evec[e * 3 + 0] * g[dst * 3 + 0]
                 + evec[e * 3 + 1] * g[dst * 3 + 1]
                 + evec[e * 3 + 2] * g[dst * 3 + 2];
        float ds = decay_scale[0];
        float sp = (ds > 20.0f) ? ds : log1pf(__expf(ds));   // softplus
        float lg = (p + et) * 0.08838834764831845f - sp * elen[e]; // 1/sqrt(128)
        logits[e] = lg;
        atomicMaxFloat(&mx[dst], lg);
    }
}

// ---------------------------------------------------------------------------
// ex = exp(logit - mx[dst]) (in place) ; den[dst] += ex
// ---------------------------------------------------------------------------
__global__ void edge_expsum_kernel(const int* __restrict__ ei,
                                   float* __restrict__ logits,
                                   const float* __restrict__ mx,
                                   float* __restrict__ den, int E)
{
    int e = blockIdx.x * blockDim.x + threadIdx.x;
    if (e >= E) return;
    int dst = ei[E + e];
    float m = mx[dst];
    if (!__builtin_isfinite(m)) m = 0.0f;
    float ex = __expf(logits[e] - m);
    logits[e] = ex;
    unsafeAtomicAdd(&den[dst], ex);
}

// ---------------------------------------------------------------------------
// attention scatter: wave per edge; out_agg[dst] += clip(alpha * v[src], +-3)
// ---------------------------------------------------------------------------
__global__ void attn_scatter_kernel(const float* __restrict__ v,
                                    const float* __restrict__ ex,
                                    const int* __restrict__ ei,
                                    const float* __restrict__ den,
                                    float* __restrict__ oagg, int E)
{
    long long t = (long long)blockIdx.x * blockDim.x + threadIdx.x;
    int e = (int)(t >> 5);
    if (e >= E) return;
    int lane = (int)(t & 31);
    int src = ei[e], dst = ei[E + e];
    float alpha = ex[e] / (den[dst] + 1e-9f);
    float4 vv = ((const float4*)(v + (size_t)src * DIM))[lane];
    float* pd = oagg + (size_t)dst * DIM + lane * 4;
    unsafeAtomicAdd(pd + 0, fminf(3.0f, fmaxf(-3.0f, alpha * vv.x)));
    unsafeAtomicAdd(pd + 1, fminf(3.0f, fmaxf(-3.0f, alpha * vv.y)));
    unsafeAtomicAdd(pd + 2, fminf(3.0f, fmaxf(-3.0f, alpha * vv.z)));
    unsafeAtomicAdd(pd + 3, fminf(3.0f, fmaxf(-3.0f, alpha * vv.w)));
}

// ---------------------------------------------------------------------------
extern "C" void kernel_launch(void* const* d_in, const int* in_sizes, int n_in,
                              void* d_out, int out_size, void* d_ws, size_t ws_size,
                              hipStream_t stream)
{
    const float* h    = (const float*)d_in[0];
    const int*   ei   = (const int*)  d_in[1];
    const float* evec = (const float*)d_in[2];
    const float* elen = (const float*)d_in[3];
    const float* Wmsg = (const float*)d_in[4];
    const float* Wupd = (const float*)d_in[5];
    const float* Wq   = (const float*)d_in[6];
    const float* Wk   = (const float*)d_in[7];
    const float* Wv   = (const float*)d_in[8];
    const float* We   = (const float*)d_in[9];
    const float* Wo   = (const float*)d_in[10];
    const float* dsc  = (const float*)d_in[11];
    float* out = (float*)d_out;

    const int N = in_sizes[0] / DIM;   // 20000
    const int E = in_sizes[3];         // 640000
    const size_t ND = (size_t)N * DIM;

    // workspace layout (floats)
    float* ws   = (float*)d_ws;
    float* hW   = ws;            // [N,128]
    float* agg  = hW   + ND;     // [N,128]
    float* h2   = agg  + ND;     // [N,128]
    float* q    = h2   + ND;     // [N,128]
    float* kf   = q    + ND;     // [N,128]
    float* vf   = kf   + ND;     // [N,128]
    float* oagg = vf   + ND;     // [N,128]
    float* g    = oagg + ND;     // [N,3]
    float* lg   = g + (size_t)N * 3;   // [E] logits -> exp
    float* mx   = lg + E;        // [N]
    float* den  = mx + N;        // [N]
    unsigned short* wbf = (unsigned short*)(den + N); // 6 x [128][128] bf16 (transposed)

    const dim3 B(256);
    const int nbRows  = (N + 127) / 128;
    const int nbND    = (int)((ND + 255) / 256);
    const int nbEdgeW = (int)(((long long)E * 32 + 255) / 256);
    const int nbEdge  = (E + 255) / 256;
    const int nbNode  = (N + 255) / 256;
    const int WSZ = DIM * DIM;   // 16384 elements per matrix

    // 0: init scratch accumulators ; convert all weights to transposed bf16 once
    init_ws_kernel<<<nbND, B, 0, stream>>>(agg, oagg, den, mx, (int)ND, N);
    weights_to_bf16t<<<(6 * WSZ + 255) / 256, B, 0, stream>>>(
        Wmsg, Wupd, Wq, Wk, Wv, Wo, wbf);

    // 1: hW = h @ W_msg  (WMMA)
    gemm128_wmma_multi<<<nbRows, B, 0, stream>>>(
        h, wbf + 0 * WSZ, nullptr, nullptr, nullptr, hW, nullptr, nullptr, N);

    // 2: agg[dst] += exp(-6*len) * hW[src]
    mp_scatter_kernel<<<nbEdgeW, B, 0, stream>>>(hW, ei, elen, agg, E);

    // 3: h2 = h + agg @ W_upd  (WMMA, fused residual)
    gemm128_wmma_multi<<<nbRows, B, 0, stream>>>(
        agg, wbf + 1 * WSZ, nullptr, nullptr, h, h2, nullptr, nullptr, N);

    // 4: q,k,v projections fused: one A stage, three weights (WMMA)
    gemm128_wmma_multi<<<nbRows, B, 0, stream>>>(
        h2, wbf + 2 * WSZ, wbf + 3 * WSZ, wbf + 4 * WSZ, nullptr, q, kf, vf, N);

    // 5: g = q @ We^T  (avoids materializing [E,128] edge keys)
    qWe_kernel<<<nbNode, B, 0, stream>>>(q, We, g, N);

    // 6: edge logits + segment max
    edge_logits_kernel<<<nbEdgeW, B, 0, stream>>>(q, kf, g, ei, evec, elen,
                                                  dsc, lg, mx, E);

    // 7: exp + segment sum
    edge_expsum_kernel<<<nbEdge, B, 0, stream>>>(ei, lg, mx, den, E);

    // 8: clipped attention scatter
    attn_scatter_kernel<<<nbEdgeW, B, 0, stream>>>(vf, lg, ei, den, oagg, E);

    // 9: out = h2 + out_agg @ Wo  (WMMA, fused residual)
    gemm128_wmma_multi<<<nbRows, B, 0, stream>>>(
        oagg, wbf + 5 * WSZ, nullptr, nullptr, h2, out, nullptr, nullptr, N);
}
